// LigerFusedNeighborhoodAttention_32066225832194
// MI455X (gfx1250) — matrix operand order repaired and checked by reading, not compiled, not run on hardware
//
#include <hip/hip_runtime.h>
#include <math.h>

// ---------------------------------------------------------------------------
// Types for CDNA5 WMMA (wave32, 16x16x32 bf16 -> f32)
// ---------------------------------------------------------------------------
typedef __attribute__((ext_vector_type(16))) __bf16 v16bf;
typedef __attribute__((ext_vector_type(8)))  __bf16 v8bf;
typedef __attribute__((ext_vector_type(4)))  __bf16 v4bf;
typedef __attribute__((ext_vector_type(8)))  float  v8f;

__device__ __forceinline__ __bf16 f2bf(float f) {
  unsigned u = __builtin_bit_cast(unsigned, f);
  unsigned r = u + 0x7fffu + ((u >> 16) & 1u);   // round-to-nearest-even
  unsigned short h = (unsigned short)(r >> 16);
  return __builtin_bit_cast(__bf16, h);
}
__device__ __forceinline__ float bf2f(__bf16 b) {
  unsigned short h = __builtin_bit_cast(unsigned short, b);
  unsigned u = ((unsigned)h) << 16;
  return __builtin_bit_cast(float, u);
}

// ---------------------------------------------------------------------------
// CDNA5 async cache->LDS copy (bypasses VGPRs; tracked by ASYNCcnt)
//   global_load_async_to_lds_b128 vdst(LDS byte addr), vaddr(64b global), off
// ---------------------------------------------------------------------------
__device__ __forceinline__ void async_copy16(void* lds_ptr, const void* gptr) {
  unsigned lds_off = (unsigned)(unsigned long long)lds_ptr; // low 32b of LDS aperture addr
  unsigned long long ga = (unsigned long long)gptr;
  asm volatile("global_load_async_to_lds_b128 %0, %1, off"
               :
               : "v"(lds_off), "v"(ga)
               : "memory");
}

__device__ __forceinline__ void wait_async0() {
#if __has_builtin(__builtin_amdgcn_s_wait_asynccnt)
  __builtin_amdgcn_s_wait_asynccnt(0);
#else
  asm volatile("s_wait_asynccnt 0x0" ::: "memory");
#endif
}

// ---------------------------------------------------------------------------
// Kernel 1: f32 -> bf16 bulk convert (hidden states), 4 elems/thread
// ---------------------------------------------------------------------------
__global__ __launch_bounds__(256)
void cvt_f32_to_bf16_kernel(const float* __restrict__ x,
                            __bf16* __restrict__ y, int n) {
  int i = (blockIdx.x * 256 + threadIdx.x) * 4;
  if (i >= n) return;
  float4 v = *(const float4*)(x + i);
  v4bf o;
  o[0] = f2bf(v.x); o[1] = f2bf(v.y); o[2] = f2bf(v.z); o[3] = f2bf(v.w);
  *(v4bf*)(y + i) = o;
}

// ---------------------------------------------------------------------------
// Kernel 2: W[k][n] f32 -> Wt[n][k] bf16 (tiled transpose through LDS)
// ---------------------------------------------------------------------------
__global__ __launch_bounds__(256)
void transpose_w_bf16_kernel(const float* __restrict__ W,
                             __bf16* __restrict__ Wt, int K, int N) {
  __shared__ float t[32][33];
  const int kt = blockIdx.x * 32;
  const int nt = blockIdx.y * 32;
  const int tx = threadIdx.x;      // 0..31
  const int ty = threadIdx.y;      // 0..7
#pragma unroll
  for (int i = 0; i < 32; i += 8)
    t[ty + i][tx] = W[(size_t)(kt + ty + i) * N + nt + tx];
  __syncthreads();
#pragma unroll
  for (int i = 0; i < 32; i += 8)
    Wt[(size_t)(nt + ty + i) * K + kt + tx] = f2bf(t[tx][ty + i]);
}

// ---------------------------------------------------------------------------
// Kernel 3: bf16 GEMM  Y[M][N] = X[M][K] * Wt[N][K]^T + bias
//   block = 256 threads (8 wave32), tile 128x128, K-step 32, double-buffered
//   LDS staging via global_load_async_to_lds_b128 (ASYNCcnt), publish with
//   s_wait_asynccnt 0 + workgroup barrier; each wave: 32x64 = 2x4 WMMA grid
// ---------------------------------------------------------------------------
template <bool OUT_F32>
__global__ __launch_bounds__(256)
void gemm_bf16_wmma_kernel(const __bf16* __restrict__ X,   // [M][K]
                           const __bf16* __restrict__ Wt,  // [N][K]
                           const float*  __restrict__ bias,// [N]
                           void* __restrict__ Yv,
                           int M, int N, int K) {
  constexpr int LDSS = 40;                    // padded row stride (bf16 elems)
  __shared__ __bf16 As[2][128 * LDSS];
  __shared__ __bf16 Bs[2][128 * LDSS];

  const int tid  = threadIdx.x;
  const int lane = tid & 31;
  const int wave = tid >> 5;
  const int wm   = wave >> 1;                 // 0..3 : 32-row band
  const int wn   = wave & 1;                  // 0..1 : 64-col band
  const int m0   = blockIdx.y * 128;
  const int n0   = blockIdx.x * 128;
  const int l16  = lane & 15;
  const int lhi  = lane >> 4;

  // async DMA stage of one 128x32 bf16 tile pair: 512 x 16B chunks each,
  // 2 chunks per thread per matrix
  auto stage = [&](int buf, int kt) {
#pragma unroll
    for (int t = 0; t < 2; ++t) {
      const int c   = tid + t * 256;
      const int row = c >> 2;
      const int ck  = c & 3;
      async_copy16(&As[buf][row * LDSS + ck * 8],
                   X  + (size_t)(m0 + row) * K + kt * 32 + ck * 8);
      async_copy16(&Bs[buf][row * LDSS + ck * 8],
                   Wt + (size_t)(n0 + row) * K + kt * 32 + ck * 8);
    }
  };

  v8f acc[2][4] = {};

  const int KT = K >> 5;
  stage(0, 0);
  wait_async0();
  __syncthreads();

  for (int kt = 0; kt < KT; ++kt) {
    const int cur = kt & 1;
    if (kt + 1 < KT) stage(cur ^ 1, kt + 1);   // in flight during compute
    if (kt + 2 < KT) {
      // speculative prefetch of the tile after next (global_prefetch_b8)
      __builtin_prefetch(X  + (size_t)(m0 + (tid >> 1)) * K + (kt + 2) * 32, 0, 1);
      __builtin_prefetch(Wt + (size_t)(n0 + (tid >> 1)) * K + (kt + 2) * 32, 0, 1);
    }

    const __bf16* A_ = &As[cur][0];
    const __bf16* B_ = &Bs[cur][0];

    // A fragments: 16-bit A 16x32 layout — lane half selects K-half
    // lanes 0-15 (row=l16): K 0..7 and 16..23 ; lanes 16-31: K 8..15 and 24..31
    v16bf a[2];
#pragma unroll
    for (int m = 0; m < 2; ++m) {
      const int row = wm * 32 + m * 16 + l16;
      const int k0  = lhi * 8;
      v8bf lo = *(const v8bf*)(A_ + row * LDSS + k0);
      v8bf hi = *(const v8bf*)(A_ + row * LDSS + k0 + 16);
      a[m] = __builtin_shufflevector(lo, hi, 0, 1, 2, 3, 4, 5, 6, 7,
                                             8, 9, 10, 11, 12, 13, 14, 15);
    }
    // B fragments: lane = N column; lanes 0-15: K 0..7 and 16..23 ; lanes 16-31: K 8..15 and 24..31
    v16bf b[4];
#pragma unroll
    for (int n = 0; n < 4; ++n) {
      const int col = wn * 64 + n * 16 + l16;
      const int k0  = lhi * 8;
      v8bf lo = *(const v8bf*)(B_ + col * LDSS + k0);
      v8bf hi = *(const v8bf*)(B_ + col * LDSS + k0 + 16);
      b[n] = __builtin_shufflevector(lo, hi, 0, 1, 2, 3, 4, 5, 6, 7,
                                             8, 9, 10, 11, 12, 13, 14, 15);
    }
#pragma unroll
    for (int m = 0; m < 2; ++m)
#pragma unroll
      for (int n = 0; n < 4; ++n)
        acc[m][n] = __builtin_amdgcn_wmma_f32_16x16x32_bf16(
            false, a[m], false, b[n], (short)0, acc[m][n], false, false);

    wait_async0();       // our async fills of the next buffer are complete
    __syncthreads();     // publish to all waves in the workgroup
  }

  // Epilogue: C/D layout — VGPR r: lanes 0-15 -> M=r, lanes 16-31 -> M=r+8
#pragma unroll
  for (int m = 0; m < 2; ++m) {
#pragma unroll
    for (int n = 0; n < 4; ++n) {
      const int rbase = m0 + wm * 32 + m * 16 + lhi * 8;
      const int col   = n0 + wn * 64 + n * 16 + l16;
      const float bv  = bias[col];
#pragma unroll
      for (int r = 0; r < 8; ++r) {
        const float v = acc[m][n][r] + bv;
        const size_t off = (size_t)(rbase + r) * N + col;
        if constexpr (OUT_F32) ((float*)Yv)[off] = v;
        else                   ((__bf16*)Yv)[off] = f2bf(v);
      }
    }
  }
}

// ---------------------------------------------------------------------------
// Kernel 4: neighborhood attention, window 7, head_dim 64
//   one thread per (b, head, s); QKV stored as [b][s][head*64+d] bf16
// ---------------------------------------------------------------------------
__global__ __launch_bounds__(256)
void neighborhood_attn_kernel(const __bf16* __restrict__ Q,
                              const __bf16* __restrict__ Kx,
                              const __bf16* __restrict__ V,
                              __bf16* __restrict__ O,
                              int S, int HEADS) {
  const int bh = blockIdx.y;                    // b*HEADS + h
  const int s  = blockIdx.x * 256 + threadIdx.x;
  const int b  = bh / HEADS;
  const int h  = bh % HEADS;
  const int HD = HEADS * 64;                    // 1024
  const size_t rowbase = (size_t)b * S * HD + (size_t)h * 64;
  const __bf16* qp = Q + rowbase + (size_t)s * HD;

  float sc[7];
#pragma unroll
  for (int j = 0; j < 7; ++j) {
    const int t = s + j - 3;
    if (t < 0 || t >= S) { sc[j] = -3.0e38f; continue; }
    const __bf16* kp = Kx + rowbase + (size_t)t * HD;
    float d = 0.f;
#pragma unroll
    for (int c = 0; c < 64; c += 8) {
      v8bf qv = *(const v8bf*)(qp + c);
      v8bf kv = *(const v8bf*)(kp + c);
#pragma unroll
      for (int i = 0; i < 8; ++i) d = fmaf(bf2f(qv[i]), bf2f(kv[i]), d);
    }
    sc[j] = d * 0.125f;                         // 1/sqrt(64)
  }

  float mx = sc[0];
#pragma unroll
  for (int j = 1; j < 7; ++j) mx = fmaxf(mx, sc[j]);
  float w[7], sum = 0.f;
#pragma unroll
  for (int j = 0; j < 7; ++j) { w[j] = __expf(sc[j] - mx); sum += w[j]; }
  const float inv = 1.0f / sum;
#pragma unroll
  for (int j = 0; j < 7; ++j) w[j] *= inv;

  __bf16* op = O + rowbase + (size_t)s * HD;
#pragma unroll
  for (int c = 0; c < 64; c += 16) {
    float o[16] = {};
#pragma unroll
    for (int j = 0; j < 7; ++j) {
      const int t = s + j - 3;
      if (t < 0 || t >= S) continue;
      const __bf16* vp = V + rowbase + (size_t)t * HD + c;
      v8bf v0 = *(const v8bf*)(vp);
      v8bf v1 = *(const v8bf*)(vp + 8);
#pragma unroll
      for (int i = 0; i < 8; ++i) {
        o[i]     = fmaf(w[j], bf2f(v0[i]), o[i]);
        o[i + 8] = fmaf(w[j], bf2f(v1[i]), o[i + 8]);
      }
    }
    v8bf r0, r1;
#pragma unroll
    for (int i = 0; i < 8; ++i) { r0[i] = f2bf(o[i]); r1[i] = f2bf(o[i + 8]); }
    *(v8bf*)(op + c)     = r0;
    *(v8bf*)(op + c + 8) = r1;
  }
}

// ---------------------------------------------------------------------------
// Host launcher
// ---------------------------------------------------------------------------
extern "C" void kernel_launch(void* const* d_in, const int* in_sizes, int n_in,
                              void* d_out, int out_size, void* d_ws, size_t ws_size,
                              hipStream_t stream) {
  constexpr int Bc = 2, Sc = 8192, Hc = 1024, HEADS = 16;
  constexpr int M = Bc * Sc;        // 16384
  constexpr int K = Hc, N = Hc;     // 1024

  const float* hidden = (const float*)d_in[0];
  const float* w_q = (const float*)d_in[1];
  const float* b_q = (const float*)d_in[2];
  const float* w_k = (const float*)d_in[3];
  const float* b_k = (const float*)d_in[4];
  const float* w_v = (const float*)d_in[5];
  const float* b_v = (const float*)d_in[6];
  const float* w_o = (const float*)d_in[7];
  const float* b_o = (const float*)d_in[8];

  // Workspace carve-out (all bf16 intermediates): ~176 MB total
  char* ws = (char*)d_ws;
  size_t off = 0;
  auto carve = [&](size_t bytes) -> void* {
    void* p = ws + off;
    off += (bytes + 255) & ~(size_t)255;
    return p;
  };
  __bf16* Xb  = (__bf16*)carve((size_t)M * K * 2);
  __bf16* Wqt = (__bf16*)carve((size_t)N * K * 2);
  __bf16* Wkt = (__bf16*)carve((size_t)N * K * 2);
  __bf16* Wvt = (__bf16*)carve((size_t)N * K * 2);
  __bf16* Wot = (__bf16*)carve((size_t)N * K * 2);
  __bf16* Qb  = (__bf16*)carve((size_t)M * N * 2);
  __bf16* Kb  = (__bf16*)carve((size_t)M * N * 2);
  __bf16* Vb  = (__bf16*)carve((size_t)M * N * 2);
  __bf16* Cb  = (__bf16*)carve((size_t)M * N * 2);

  // 1) hidden f32 -> bf16
  cvt_f32_to_bf16_kernel<<<(M * K) / 1024, 256, 0, stream>>>(hidden, Xb, M * K);

  // 2) weight transpose+convert: W[k][n] f32 -> Wt[n][k] bf16
  dim3 tb(32, 8);
  dim3 tg(K / 32, N / 32);
  transpose_w_bf16_kernel<<<tg, tb, 0, stream>>>(w_q, Wqt, K, N);
  transpose_w_bf16_kernel<<<tg, tb, 0, stream>>>(w_k, Wkt, K, N);
  transpose_w_bf16_kernel<<<tg, tb, 0, stream>>>(w_v, Wvt, K, N);
  transpose_w_bf16_kernel<<<tg, tb, 0, stream>>>(w_o, Wot, K, N);

  // 3) QKV projections (WMMA bf16)
  dim3 gg(N / 128, M / 128);
  gemm_bf16_wmma_kernel<false><<<gg, 256, 0, stream>>>(Xb, Wqt, b_q, Qb, M, N, K);
  gemm_bf16_wmma_kernel<false><<<gg, 256, 0, stream>>>(Xb, Wkt, b_k, Kb, M, N, K);
  gemm_bf16_wmma_kernel<false><<<gg, 256, 0, stream>>>(Xb, Wvt, b_v, Vb, M, N, K);

  // 4) neighborhood attention
  dim3 ag(Sc / 256, Bc * HEADS);
  neighborhood_attn_kernel<<<ag, 256, 0, stream>>>(Qb, Kb, Vb, Cb, Sc, HEADS);

  // 5) output projection -> f32 d_out
  gemm_bf16_wmma_kernel<true><<<gg, 256, 0, stream>>>(Cb, Wot, b_o, (float*)d_out, M, N, K);
}
